// EulerInteractionLayer_12180527251618
// MI455X (gfx1250) — compile-verified, matrix operand order checked by reading, not run on hardware
//
#include <hip/hip_runtime.h>

// EulerInteractionLayer for MI455X (gfx1250, wave32, WMMA, async-to-LDS).
// B=8192, F=50, D=64, K=N=3200.

typedef __attribute__((ext_vector_type(16))) _Float16 v16h;
typedef __attribute__((ext_vector_type(8)))  float    v8f;
typedef __attribute__((ext_vector_type(4)))  int      v4i;

#define BATCH 8192
#define NF    50
#define ND    64
#define KDIM  3200   // = NF * ND

#define TM    128    // GEMM M tile
#define TN    128    // GEMM N tile = 2 LayerNorm groups of 64
#define AST   40     // LDS row stride in halves (80B: 16B-aligned, conflict-spread)
#define BST   40
#define CSTR  129    // epilogue Cs row stride in floats (odd -> conflict-free stats)
#define NBUF  3      // async pipeline depth

#define AS1 __attribute__((address_space(1)))
#define AS3 __attribute__((address_space(3)))

#if __has_builtin(__builtin_amdgcn_global_load_async_to_lds_b128) && \
    __has_builtin(__builtin_amdgcn_s_wait_asynccnt)
#define USE_ASYNC_LDS 1
#else
#define USE_ASYNC_LDS 0
#endif

// ---------------------------------------------------------------------------
// Transpose + cast im_w [K][N] f32 -> Wt [N][K] f16 (32x32 LDS tiles)
// ---------------------------------------------------------------------------
__global__ __launch_bounds__(256) void transpose_to_half(const float* __restrict__ w,
                                                         _Float16* __restrict__ wt) {
  __shared__ float t[32][33];
  const int bx = blockIdx.x * 32;           // input row base (k)
  const int by = blockIdx.y * 32;           // input col base (n)
  const int tx = threadIdx.x & 31;
  const int ty = threadIdx.x >> 5;          // 0..7
#pragma unroll
  for (int i = 0; i < 4; ++i) {
    const int row = ty + i * 8;
    t[row][tx] = w[(size_t)(bx + row) * KDIM + by + tx];
  }
  __syncthreads();
#pragma unroll
  for (int i = 0; i < 4; ++i) {
    const int row = ty + i * 8;
    wt[(size_t)(by + row) * KDIM + bx + tx] = (_Float16)t[tx][row];
  }
}

// ---------------------------------------------------------------------------
// Polar branch + f16 casts of the activations (fused to save a read pass).
//   er[b,f,d] = exp(lam_t[b,d,f]) * cos(theta_t[b,d,f])
//   ep[b,f,d] = exp(lam_t[b,d,f]) * sin(theta_t[b,d,f])
// ---------------------------------------------------------------------------
__global__ __launch_bounds__(256) void euler_precompute(
    const float* __restrict__ r, const float* __restrict__ p,
    const float* __restrict__ M,                    // inter_orders [F][F]
    const float* __restrict__ bl,                   // bias_lam  [D][F]
    const float* __restrict__ bt,                   // bias_theta[D][F]
    float* __restrict__ er, float* __restrict__ ep,
    _Float16* __restrict__ rh, _Float16* __restrict__ ph) {
  __shared__ float lam_s[KDIM];
  __shared__ float th_s[KDIM];
  __shared__ float M_s[NF * NF];
  __shared__ float bl_s[KDIM];
  __shared__ float bt_s[KDIM];

  const int b = blockIdx.x;
  const int tid = threadIdx.x;
  const size_t base = (size_t)b * KDIM;

  for (int i = tid; i < KDIM; i += 256) {
    const float rv = r[base + i];
    const float pv = p[base + i];
    rh[base + i] = (_Float16)rv;                    // fused f32->f16 cast
    ph[base + i] = (_Float16)pv;
    lam_s[i] = 0.5f * __logf(rv * rv + pv * pv + 1e-8f);
    th_s[i]  = atan2f(pv, rv);
    bl_s[i]  = bl[i];
    bt_s[i]  = bt[i];
  }
  for (int i = tid; i < NF * NF; i += 256) M_s[i] = M[i];
  __syncthreads();

  for (int idx = tid; idx < KDIM; idx += 256) {
    const int d  = idx & 63;
    const int fo = idx >> 6;                        // output field
    float sl = bl_s[d * NF + fo];
    float st = bt_s[d * NF + fo];
#pragma unroll 10
    for (int f = 0; f < NF; ++f) {
      const float m = M_s[f * NF + fo];
      sl = fmaf(lam_s[f * ND + d], m, sl);
      st = fmaf(th_s[f * ND + d], m, st);
    }
    const float e = __expf(sl);
    float s, c;
    __sincosf(st, &s, &c);
    er[base + idx] = e * c;
    ep[base + idx] = e * s;
  }
}

// ---------------------------------------------------------------------------
// Fused GEMM + bias + ReLU + euler add + LayerNorm(D=64).
// 128x128 block tile, 8 wave32 each owning 32x64 (2 A-frags x 4 B-frags =
// 8 v_wmma_f32_16x16x32_f16 per K-step of 32; K = 3200).
// 3-deep async global->LDS pipeline: each tile's loads stay in flight for a
// full K-step (s_wait_asynccnt 4), only the last tile waits to zero.
// ---------------------------------------------------------------------------
struct GemmBufs {
  _Float16 A[NBUF][TM * AST];   // 3 x 10240 B
  _Float16 B[NBUF][TN * BST];   // 3 x 10240 B
};
struct EpiBufs {
  float C[TM * CSTR];           // 66048 B
  float mu[256];
  float rs[256];
};

__global__ __launch_bounds__(256) void gemm_fused(
    const _Float16* __restrict__ X,     // [B][K] f16
    const _Float16* __restrict__ Wt,    // [N][K] f16 (pre-transposed weights)
    const float* __restrict__ bias,     // [N]
    const float* __restrict__ euler,    // [B][N] f32 (lam*cos or lam*sin)
    const float* __restrict__ lnw, const float* __restrict__ lnb,  // [D]
    float* __restrict__ out) {
  __shared__ union { GemmBufs g; EpiBufs e; } sm;

  const int tid  = threadIdx.x;
  const int wave = tid >> 5;            // 0..7 (wave32)
  const int lane = tid & 31;
  const int wave_m = wave & 3;          // row group: 32 rows each
  const int wave_n = wave >> 2;         // col group: 64 cols each
  const int bm = blockIdx.x * TM;
  const int bn = blockIdx.y * TN;

  v8f acc[2][4];
#pragma unroll
  for (int mi = 0; mi < 2; ++mi)
#pragma unroll
    for (int ni = 0; ni < 4; ++ni)
      acc[mi][ni] = v8f{0.f, 0.f, 0.f, 0.f, 0.f, 0.f, 0.f, 0.f};

  // Cooperative tile-load mapping: A and B are both 128 rows x 32 halves,
  // 16 halves (2 x b128) per thread.
  const int arow = tid >> 1;
  const int aoff = (tid & 1) << 4;
  const _Float16* gA = X + (size_t)(bm + arow) * KDIM + aoff;
  const _Float16* gB = Wt + (size_t)(bn + arow) * KDIM + aoff;

  // WMMA fragment addressing (ISA 7.12.2, 16-bit A 16x32):
  // lane<16: K = {0..7, 16..23} of row (lane&15); lane>=16: K = {8..15, 24..31}
  const int fr  = lane & 15;
  const int fks = (lane >> 4) << 3;

  union F16 { v16h h; uint4 q[2]; };

  // 8 WMMAs on buffer `buf`
  auto compute = [&](int buf) {
    F16 a[2], b[4];
#pragma unroll
    for (int mi = 0; mi < 2; ++mi) {
      const _Float16* ap = &sm.g.A[buf][(wave_m * 32 + mi * 16 + fr) * AST + fks];
      a[mi].q[0] = *(const uint4*)(ap);
      a[mi].q[1] = *(const uint4*)(ap + 16);
    }
#pragma unroll
    for (int ni = 0; ni < 4; ++ni) {
      const _Float16* bp = &sm.g.B[buf][(wave_n * 64 + ni * 16 + fr) * BST + fks];
      b[ni].q[0] = *(const uint4*)(bp);
      b[ni].q[1] = *(const uint4*)(bp + 16);
    }
#pragma unroll
    for (int mi = 0; mi < 2; ++mi)
#pragma unroll
      for (int ni = 0; ni < 4; ++ni)
        acc[mi][ni] = __builtin_amdgcn_wmma_f32_16x16x32_f16(
            false, a[mi].h, false, b[ni].h, (short)0, acc[mi][ni], false, false);
  };

  const int NSTEP = KDIM / 32;          // 100

#if USE_ASYNC_LDS
  // 4 async b128 per thread per tile, tracked with ASYNCcnt.
  auto load_tile = [&](int buf, int kk) {
    __builtin_amdgcn_global_load_async_to_lds_b128(
        (AS1 v4i*)(gA + kk),     (AS3 v4i*)&sm.g.A[buf][arow * AST + aoff], 0, 0);
    __builtin_amdgcn_global_load_async_to_lds_b128(
        (AS1 v4i*)(gA + kk + 8), (AS3 v4i*)&sm.g.A[buf][arow * AST + aoff + 8], 0, 0);
    __builtin_amdgcn_global_load_async_to_lds_b128(
        (AS1 v4i*)(gB + kk),     (AS3 v4i*)&sm.g.B[buf][arow * AST + aoff], 0, 0);
    __builtin_amdgcn_global_load_async_to_lds_b128(
        (AS1 v4i*)(gB + kk + 8), (AS3 v4i*)&sm.g.B[buf][arow * AST + aoff + 8], 0, 0);
  };

  load_tile(0, 0);
  load_tile(1, 32);
  int cur = 0, nb = 2;
  for (int kt = 0; kt < NSTEP - 1; ++kt) {
    __builtin_amdgcn_s_wait_asynccnt(4);   // tile kt landed; tile kt+1 may fly
    __syncthreads();                       // all waves' tile-kt writes visible
    if (kt + 2 < NSTEP) load_tile(nb, (kt + 2) * 32);
    compute(cur);
    cur = (cur == NBUF - 1) ? 0 : cur + 1;
    nb  = (nb  == NBUF - 1) ? 0 : nb + 1;
  }
  __builtin_amdgcn_s_wait_asynccnt(0);
  __syncthreads();
  compute(cur);
#else
  // Fallback: global -> VGPR -> LDS double buffer.
  auto load_regs = [&](int kk, uint4& qa0, uint4& qa1, uint4& qb0, uint4& qb1) {
    const uint4* g = (const uint4*)(gA + kk);
    qa0 = g[0]; qa1 = g[1];
    const uint4* h = (const uint4*)(gB + kk);
    qb0 = h[0]; qb1 = h[1];
  };
  auto store_tile = [&](int buf, const uint4& qa0, const uint4& qa1,
                        const uint4& qb0, const uint4& qb1) {
    uint4* la = (uint4*)&sm.g.A[buf][arow * AST + aoff];
    la[0] = qa0; la[1] = qa1;
    uint4* lb = (uint4*)&sm.g.B[buf][arow * AST + aoff];
    lb[0] = qb0; lb[1] = qb1;
  };
  {
    uint4 qa0, qa1, qb0, qb1;
    load_regs(0, qa0, qa1, qb0, qb1);
    store_tile(0, qa0, qa1, qb0, qb1);
  }
  __syncthreads();
  for (int kt = 0; kt < NSTEP; ++kt) {
    uint4 qa0, qa1, qb0, qb1;
    const bool more = (kt + 1 < NSTEP);
    if (more) load_regs((kt + 1) * 32, qa0, qa1, qb0, qb1);
    compute(kt & 1);
    if (more) store_tile((kt + 1) & 1, qa0, qa1, qb0, qb1);
    __syncthreads();
  }
#endif

  __syncthreads();   // Cs aliases the tile buffers: all LDS reads must be done

  // ---- Epilogue: bias + ReLU + euler -> Cs ---------------------------------
  // C/D layout: VGPR i: lanes 0-15 -> M=i, N=lane; lanes 16-31 -> M=8+i, N=lane-16
  const int rbase = wave_m * 32 + ((lane >> 4) << 3);
  const int ccol  = lane & 15;
#pragma unroll
  for (int mi = 0; mi < 2; ++mi) {
#pragma unroll
    for (int ni = 0; ni < 4; ++ni) {
      const int ncol = wave_n * 64 + ni * 16 + ccol;
      const float bz = bias[bn + ncol];
#pragma unroll
      for (int i = 0; i < 8; ++i) {
        const int row = rbase + mi * 16 + i;
        float v = acc[mi][ni][i] + bz;
        v = fmaxf(v, 0.0f);
        v += euler[(size_t)(bm + row) * KDIM + bn + ncol];
        sm.e.C[row * CSTR + ncol] = v;
      }
    }
  }
  __syncthreads();

  // ---- LayerNorm stats: 256 (row, group) pairs, one per thread -------------
  {
    const int row = tid & 127;
    const int grp = tid >> 7;
    float s = 0.f, s2 = 0.f;
#pragma unroll
    for (int j = 0; j < 64; ++j) {
      const float v = sm.e.C[row * CSTR + grp * 64 + j];
      s += v; s2 += v * v;
    }
    const float mu  = s * (1.0f / 64.0f);
    const float var = s2 * (1.0f / 64.0f) - mu * mu;
    sm.e.mu[tid] = mu;
    sm.e.rs[tid] = rsqrtf(var + 1e-5f);
  }
  __syncthreads();

  // ---- normalize + store ---------------------------------------------------
#pragma unroll
  for (int i = 0; i < (TM * TN) / 256; ++i) {
    const int e = i * 256 + tid;
    const int row = e >> 7;
    const int col = e & 127;
    const int d   = col & 63;
    const int sg  = (col >> 6) * 128 + row;
    const float v = (sm.e.C[row * CSTR + col] - sm.e.mu[sg]) * sm.e.rs[sg] * lnw[d] + lnb[d];
    out[(size_t)(bm + row) * KDIM + bn + col] = v;
  }
}

// ---------------------------------------------------------------------------
extern "C" void kernel_launch(void* const* d_in, const int* in_sizes, int n_in,
                              void* d_out, int out_size, void* d_ws, size_t ws_size,
                              hipStream_t stream) {
  (void)in_sizes; (void)n_in; (void)out_size; (void)ws_size;
  const float* r   = (const float*)d_in[0];
  const float* p   = (const float*)d_in[1];
  const float* M   = (const float*)d_in[2];   // inter_orders [F,F]
  const float* w   = (const float*)d_in[3];   // im_w [K,N]
  const float* imb = (const float*)d_in[4];   // im_b [N]
  const float* bl  = (const float*)d_in[5];   // bias_lam   [1,D,F]
  const float* bt  = (const float*)d_in[6];   // bias_theta [1,D,F]
  const float* nrw = (const float*)d_in[7];
  const float* nrb = (const float*)d_in[8];
  const float* npw = (const float*)d_in[9];
  const float* npb = (const float*)d_in[10];
  float* out = (float*)d_out;

  char* ws = (char*)d_ws;
  size_t off = 0;
  auto take = [&](size_t bytes) -> char* {
    char* ptr = ws + off;
    off = (off + bytes + 255) & ~(size_t)255;
    return ptr;
  };
  _Float16* rh = (_Float16*)take((size_t)BATCH * KDIM * 2);
  _Float16* ph = (_Float16*)take((size_t)BATCH * KDIM * 2);
  _Float16* wt = (_Float16*)take((size_t)KDIM * KDIM * 2);
  float*    er = (float*)take((size_t)BATCH * KDIM * 4);
  float*    ep = (float*)take((size_t)BATCH * KDIM * 4);

  transpose_to_half<<<dim3(KDIM / 32, KDIM / 32), 256, 0, stream>>>(w, wt);
  euler_precompute<<<BATCH, 256, 0, stream>>>(r, p, M, bl, bt, er, ep, rh, ph);

  const dim3 g(BATCH / TM, KDIM / TN);   // 64 x 25
  gemm_fused<<<g, 256, 0, stream>>>(rh, wt, imb, er, nrw, nrb, out);
  gemm_fused<<<g, 256, 0, stream>>>(ph, wt, imb, ep, npw, npb,
                                    out + (size_t)BATCH * KDIM);
}
